// MF2Demo_67843303407889
// MI455X (gfx1250) — matrix operand order repaired and checked by reading, not compiled, not run on hardware
//
#include <hip/hip_runtime.h>
#include <hip/hip_bf16.h>
#include <math.h>

// ---------------------------------------------------------------------------
// Problem constants (match reference)
// ---------------------------------------------------------------------------
#define Bsz   131072
#define Ddim  300
#define Knegs 32
#define Ccb   384
#define Aattr 18

#define H1    64
#define H2    32
#define D_PAD 320              // 10 K-tiles of 32 for layer 1

// LDS pitches (bf16 elements). Multiples of 8 -> 16B-aligned ds_load_b128.
#define PITCHX 328
#define PITCH1 72
#define PITCH2 40
#define PITCHW 20              // f32 W_compact tile pitch

typedef __bf16 bf16_t;
typedef __attribute__((ext_vector_type(16))) __bf16 v16bf;
typedef __attribute__((ext_vector_type(8)))  __bf16 v8bf;
typedef __attribute__((ext_vector_type(4)))  __bf16 v4bf;
typedef __attribute__((ext_vector_type(8)))  float  v8f;

// ws layout (bf16 element offsets for weight fragments)
#define W1_NKT 10
#define W1_NNT 4
#define W2_NKT 2
#define W2_NNT 2
#define W3_NNT 2
#define W2_OFF (W1_NKT * W1_NNT * 512)          // 20480 bf16 elems
#define W3_OFF (W2_OFF + W2_NKT * W2_NNT * 512) // +2048
#define FRAG_ELEMS (W3_OFF + W3_NNT * 512)      // 23552 bf16 elems
#define ACC_OFF_BYTES (((FRAG_ELEMS * 2) + 255) & ~255) // 47104

__device__ __forceinline__ float fast_sigmoid(float z) {
    return 1.0f / (1.0f + __expf(-z));
}
// numerically stable log(sigmoid(z)) = min(z,0) - log(1+exp(-|z|))
__device__ __forceinline__ float log_sigmoid(float z) {
    return fminf(z, 0.0f) - __logf(1.0f + __expf(-fabsf(z)));
}

// ---------------------------------------------------------------------------
// Kernel 0: repack W1/W2/W3 into bf16 WMMA B-fragment layout in d_ws.
// Fragment = 512 bf16; lane L holds (Khalf = L>>4, N = L&15), 16 contiguous
// bf16 covering K = ktile*32 + Khalf*16 + [0..15].
// One block (32 threads) per fragment. Block 0 also zeroes the accumulators.
// ---------------------------------------------------------------------------
__global__ void prep_kernel(const float* __restrict__ W1,
                            const float* __restrict__ W2,
                            const float* __restrict__ W3,
                            bf16_t* __restrict__ frags,
                            float* __restrict__ acc) {
    const int blk   = blockIdx.x;
    const int lane  = threadIdx.x;
    const int n     = lane & 15;
    const int khalf = lane >> 4;
    if (blk == 0 && lane < 2) acc[lane] = 0.0f;

    if (blk < W1_NKT * W1_NNT) {                       // W1: 10 kt x 4 nt
        const int kt = blk >> 2, nt = blk & 3;
        bf16_t* dst = frags + (size_t)blk * 512 + lane * 16;
#pragma unroll
        for (int e = 0; e < 16; ++e) {
            const int k = kt * 32 + khalf * 16 + e;
            const int c = nt * 16 + n;
            const float v = (k < Ddim) ? W1[(size_t)k * H1 + c] : 0.0f;
            dst[e] = (bf16_t)v;
        }
    } else if (blk < W1_NKT * W1_NNT + W2_NKT * W2_NNT) { // W2: 2 kt x 2 nt
        const int f  = blk - W1_NKT * W1_NNT;
        const int kt = f >> 1, nt = f & 1;
        bf16_t* dst = frags + (size_t)W2_OFF + (size_t)f * 512 + lane * 16;
#pragma unroll
        for (int e = 0; e < 16; ++e) {
            const int k = kt * 32 + khalf * 16 + e;
            const int c = nt * 16 + n;
            dst[e] = (bf16_t)W2[(size_t)k * H2 + c];
        }
    } else {                                            // W3: 1 kt x 2 nt (N padded 18->32)
        const int nt = blk - (W1_NKT * W1_NNT + W2_NKT * W2_NNT);
        bf16_t* dst = frags + (size_t)W3_OFF + (size_t)nt * 512 + lane * 16;
#pragma unroll
        for (int e = 0; e < 16; ++e) {
            const int k = khalf * 16 + e;
            const int c = nt * 16 + n;
            const float v = (c < Aattr) ? W3[(size_t)k * Aattr + c] : 0.0f;
            dst[e] = (bf16_t)v;
        }
    }
}

// ---------------------------------------------------------------------------
// Kernel 1: fused MLP (WMMA bf16) + masking + neg-sampling loss partials.
// 128 threads = 4 waves; each wave owns a 16-row tile (64 rows / block).
// ---------------------------------------------------------------------------
__global__ __launch_bounds__(128) void mf2_main_kernel(
    const float* __restrict__ x,   const float* __restrict__ y,
    const float* __restrict__ ob,  const float* __restrict__ cb,
    const float* __restrict__ b1,  const float* __restrict__ b2,
    const float* __restrict__ b3,  const int*   __restrict__ neg_idx,
    const bf16_t* __restrict__ frags,
    float* __restrict__ wuser_out, float* __restrict__ acc) {

    __shared__ __align__(16) bf16_t lds_x[64 * PITCHX];        // 41984 B
    __shared__ __align__(16) bf16_t lds_h1[4][16 * PITCH1];    //  9216 B
    __shared__ __align__(16) bf16_t lds_h2[4][16 * PITCH2];    //  5120 B
    __shared__ __align__(16) float  lds_wc[4][16 * PITCHW];    //  5120 B

    const int tid  = threadIdx.x;
    const int wave = tid >> 5;
    const int lane = tid & 31;
    const int n    = lane & 15;
    const int hi   = lane >> 4;
    const int row0 = blockIdx.x * 64;

    // ---- stage x tile: 64 rows x 300 f32 -> bf16 LDS ----------------------
    // 64*300 region is contiguous in global; 300 % 4 == 0, so float4 loads
    // never straddle rows. 4800 float4 total; packed 4x bf16 b64 LDS stores.
    {
        const float4* x4 = (const float4*)(x + (size_t)row0 * Ddim);
        for (int i = tid; i < (64 * Ddim) / 4; i += 128) {
            const int r  = i / (Ddim / 4);
            const int c4 = i - r * (Ddim / 4);
            const float4 v = x4[i];
            v4bf p;
            p[0] = (bf16_t)v.x; p[1] = (bf16_t)v.y;
            p[2] = (bf16_t)v.z; p[3] = (bf16_t)v.w;
            *(v4bf*)&lds_x[r * PITCHX + c4 * 4] = p;
        }
        // zero K-pad cols 300..319 (320 % 4 == 0 -> b64 stores)
        const v4bf z4 = {};
        for (int i = tid; i < (64 * (D_PAD - Ddim)) / 4; i += 128) {
            const int r  = i / ((D_PAD - Ddim) / 4);
            const int c4 = i - r * ((D_PAD - Ddim) / 4);
            *(v4bf*)&lds_x[r * PITCHX + Ddim + c4 * 4] = z4;
        }
    }
    __syncthreads();

    const v8f vzero = {};

    // ---- layer 1: [16x320] @ [320x64] via 10x4 WMMA ------------------------
    v8f acc1[4];
#pragma unroll
    for (int nt = 0; nt < 4; ++nt) acc1[nt] = vzero;

    const bf16_t* arow = &lds_x[(wave * 16 + n) * PITCHX];
#pragma unroll 2
    for (int kt = 0; kt < 10; ++kt) {
        const bf16_t* pa = arow + kt * 32 + hi * 16;
        const v8bf alo = *(const v8bf*)pa;
        const v8bf ahi = *(const v8bf*)(pa + 8);
        v16bf a;
#pragma unroll
        for (int i = 0; i < 8; ++i) { a[i] = alo[i]; a[8 + i] = ahi[i]; }
#pragma unroll
        for (int nt = 0; nt < 4; ++nt) {
            const v16bf b = *(const v16bf*)(frags + (size_t)(kt * 4 + nt) * 512 + lane * 16);
            acc1[nt] = __builtin_amdgcn_wmma_f32_16x16x32_bf16(
                false, a, false, b, (short)0, acc1[nt], false, false);
        }
    }

    // sigmoid(+b1) -> h1 bf16 in LDS
#pragma unroll
    for (int nt = 0; nt < 4; ++nt) {
        const float bias = b1[nt * 16 + n];
#pragma unroll
        for (int r = 0; r < 8; ++r) {
            const float s = fast_sigmoid(acc1[nt][r] + bias);
            lds_h1[wave][(r + hi * 8) * PITCH1 + nt * 16 + n] = (bf16_t)s;
        }
    }
    asm volatile("s_wait_dscnt 0" ::: "memory");   // intra-wave LDS RAW

    // ---- layer 2: [16x64] @ [64x32] via 2x2 WMMA ---------------------------
    v8f acc2[2]; acc2[0] = vzero; acc2[1] = vzero;
    const bf16_t* h1row = &lds_h1[wave][n * PITCH1];
#pragma unroll
    for (int kt = 0; kt < 2; ++kt) {
        const bf16_t* pa = h1row + kt * 32 + hi * 16;
        const v8bf alo = *(const v8bf*)pa;
        const v8bf ahi = *(const v8bf*)(pa + 8);
        v16bf a;
#pragma unroll
        for (int i = 0; i < 8; ++i) { a[i] = alo[i]; a[8 + i] = ahi[i]; }
#pragma unroll
        for (int nt = 0; nt < 2; ++nt) {
            const v16bf b = *(const v16bf*)(frags + (size_t)W2_OFF +
                                            (size_t)(kt * 2 + nt) * 512 + lane * 16);
            acc2[nt] = __builtin_amdgcn_wmma_f32_16x16x32_bf16(
                false, a, false, b, (short)0, acc2[nt], false, false);
        }
    }
#pragma unroll
    for (int nt = 0; nt < 2; ++nt) {
        const float bias = b2[nt * 16 + n];
#pragma unroll
        for (int r = 0; r < 8; ++r) {
            const float s = fast_sigmoid(acc2[nt][r] + bias);
            lds_h2[wave][(r + hi * 8) * PITCH2 + nt * 16 + n] = (bf16_t)s;
        }
    }
    asm volatile("s_wait_dscnt 0" ::: "memory");

    // ---- layer 3: [16x32] @ [32x32pad] via 2 WMMA --------------------------
    v8f acc3[2]; acc3[0] = vzero; acc3[1] = vzero;
    {
        const bf16_t* pa = &lds_h2[wave][n * PITCH2] + hi * 16;
        const v8bf alo = *(const v8bf*)pa;
        const v8bf ahi = *(const v8bf*)(pa + 8);
        v16bf a;
#pragma unroll
        for (int i = 0; i < 8; ++i) { a[i] = alo[i]; a[8 + i] = ahi[i]; }
#pragma unroll
        for (int nt = 0; nt < 2; ++nt) {
            const v16bf b = *(const v16bf*)(frags + (size_t)W3_OFF +
                                            (size_t)nt * 512 + lane * 16);
            acc3[nt] = __builtin_amdgcn_wmma_f32_16x16x32_bf16(
                false, a, false, b, (short)0, acc3[nt], false, false);
        }
    }

    // W_user write-out + W_compact staging (C/D layout: lane=N, VGPR r -> M)
#pragma unroll
    for (int nt = 0; nt < 2; ++nt) {
        const int col = nt * 16 + n;
        const float bias = (col < Aattr) ? b3[col] : 0.0f;
#pragma unroll
        for (int r = 0; r < 8; ++r) {
            const int grow = row0 + wave * 16 + r + hi * 8;
            const float wu = acc3[nt][r] + bias;
            if (col < Aattr) {
                wuser_out[(size_t)grow * Aattr + col] = wu;
                lds_wc[wave][(r + hi * 8) * PITCHW + col] =
                    wu * ob[(size_t)grow * Aattr + col];
            }
        }
    }
    asm volatile("s_wait_dscnt 0" ::: "memory");

    // ---- tail: per-row log-sigmoid loss, 2 lanes per row -------------------
    const int rr   = n;                       // row within wave tile
    const int grow = row0 + wave * 16 + rr;

    // pull this row's W_compact into registers (18 VGPRs)
    float wreg[Aattr];
    {
        const float* wc = &lds_wc[wave][rr * PITCHW];
#pragma unroll
        for (int a = 0; a < Aattr; ++a) wreg[a] = wc[a];
    }

    // pos_log + row-sum with b64 y loads (18 f32 = 9 float2, 8B-aligned)
    float pos = 0.0f, wsum = 0.0f;
    {
        const float2* y2 = (const float2*)(y + (size_t)grow * Aattr);
#pragma unroll
        for (int a2 = 0; a2 < Aattr / 2; ++a2) {
            const float2 yv = y2[a2];
            const float w0 = wreg[2 * a2], w1 = wreg[2 * a2 + 1];
            wsum += w0 + w1;
            pos  += log_sigmoid(w0 * yv.x) + log_sigmoid(w1 * yv.y);
        }
    }

    // negatives: each half-wave handles 16; indices via 4x int4 (16B-aligned),
    // codebook rows via 9x float2 (8B-aligned, L2-resident)
    float nsum = 0.0f;
    {
        const int4* nidx4 = (const int4*)(neg_idx + (size_t)grow * Knegs + hi * 16);
#pragma unroll
        for (int kq = 0; kq < 4; ++kq) {
            const int4 iv = nidx4[kq];
            const int ids[4] = {iv.x, iv.y, iv.z, iv.w};
#pragma unroll
            for (int j = 0; j < 4; ++j) {
                const float2* c2 = (const float2*)(cb + (size_t)ids[j] * Aattr);
#pragma unroll
                for (int a2 = 0; a2 < Aattr / 2; ++a2) {
                    const float2 cv = c2[a2];
                    nsum += log_sigmoid(-cv.x * wreg[2 * a2]);
                    nsum += log_sigmoid(-cv.y * wreg[2 * a2 + 1]);
                }
            }
        }
    }
    nsum += __shfl_xor(nsum, 16, 32);         // combine the two halves

    const float m = (wsum != 0.0f) ? 1.0f : 0.0f;
    float contrib = m * (pos + nsum);
    float cnt = m;
#pragma unroll
    for (int off = 1; off < 16; off <<= 1) {  // reduce over the 16 rows
        contrib += __shfl_xor(contrib, off, 32);
        cnt     += __shfl_xor(cnt,     off, 32);
    }
    if (lane == 0) {
        atomicAdd(&acc[0], contrib);
        atomicAdd(&acc[1], cnt);
    }
}

// ---------------------------------------------------------------------------
// Kernel 2: finalize loss
// ---------------------------------------------------------------------------
__global__ void finalize_kernel(const float* __restrict__ acc,
                                float* __restrict__ out_loss) {
    *out_loss = -acc[0] / fmaxf(acc[1], 1.0f);
}

// ---------------------------------------------------------------------------
extern "C" void kernel_launch(void* const* d_in, const int* in_sizes, int n_in,
                              void* d_out, int out_size, void* d_ws, size_t ws_size,
                              hipStream_t stream) {
    (void)in_sizes; (void)n_in; (void)out_size; (void)ws_size;
    const float* x  = (const float*)d_in[0];
    const float* y  = (const float*)d_in[1];
    const float* ob = (const float*)d_in[2];
    const float* cb = (const float*)d_in[3];
    const float* W1 = (const float*)d_in[4];
    const float* b1 = (const float*)d_in[5];
    const float* W2 = (const float*)d_in[6];
    const float* b2 = (const float*)d_in[7];
    const float* W3 = (const float*)d_in[8];
    const float* b3 = (const float*)d_in[9];
    const int*  neg = (const int*)d_in[10];

    float*  out   = (float*)d_out;                 // [B*A] W_user, then loss
    bf16_t* frags = (bf16_t*)d_ws;
    float*  acc   = (float*)((char*)d_ws + ACC_OFF_BYTES);

    const int nfrag_blocks = W1_NKT * W1_NNT + W2_NKT * W2_NNT + W3_NNT; // 46
    prep_kernel<<<nfrag_blocks, 32, 0, stream>>>(W1, W2, W3, frags, acc);
    mf2_main_kernel<<<Bsz / 64, 128, 0, stream>>>(x, y, ob, cb, b1, b2, b3,
                                                  neg, frags, out, acc);
    finalize_kernel<<<1, 1, 0, stream>>>(acc, out + (size_t)Bsz * Aattr);
}
